// MultiHeadAttention_52269751993015
// MI455X (gfx1250) — matrix operand order repaired and checked by reading, not compiled, not run on hardware
//
#include <hip/hip_runtime.h>
#include <hip/hip_bf16.h>
#include <cstddef>
#include <cstdint>

// ---------------------------------------------------------------------------
// MultiHeadAttention forward for MI455X (gfx1250, wave32, WMMA).
//   B=4, T=2048, D=1024, H=16, DH=64.
// Pipeline (all f16 WMMA with f32 accumulate):
//   1) cast x -> f16, repack Wq/Wk/Wv to n-major f16, cast Wo -> f16
//   2) GEMM (v_wmma_f32_16x16x32_f16): Q (pre-scaled by 1/sqrt(DH)*log2e),
//      K -> (B,H,T,DH); V -> (B,H,DH,T)
//   3) flash attention: unmasked main loop + single masked tail block,
//      online softmax in log2 domain, -> O f16 (B*T, D)
//   4) GEMM: out = O @ Wo^T + bo  (fp32)
// ---------------------------------------------------------------------------

#define ATTN_B 4
#define ATTN_T 2048
#define ATTN_D 1024
#define ATTN_H 16
#define ATTN_DH 64
#define ATTN_BT (ATTN_B * ATTN_T)   // 8192
#define LOG2E 1.4426950408889634f
#define NEG_BIG (-3.0e38f)

typedef __attribute__((ext_vector_type(16))) _Float16 v16h;
typedef __attribute__((ext_vector_type(8)))  float    v8f;
typedef __attribute__((ext_vector_type(8)))  unsigned int v8u;

// K-offset inside a 32-wide chunk for 16-bit A fragments (ISA 7.12.2):
// lanes 0-15: VGPR0-3 -> K 0..7, VGPR4-7 -> K 16..23; lanes 16-31: +8.
__device__ __forceinline__ int koffA(int j, int lh) {
  return (j < 4) ? (lh * 8 + 2 * j) : (16 + lh * 8 + 2 * (j - 4));
}
// 16-bit B fragment (32x16): same K order as A (VGPR0-3 -> K 0..7, VGPR4-7 -> K 16..23; lanes 16-31: +8).
__device__ __forceinline__ int koffB(int j, int lh) {
  return (j < 4) ? (lh * 8 + 2 * j) : (16 + lh * 8 + 2 * (j - 4));
}

// ---------------------------------------------------------------------------
// elementwise converts
// ---------------------------------------------------------------------------
__global__ void cvt_f32_to_f16(const float* __restrict__ in,
                               _Float16* __restrict__ out, int n) {
  int i = blockIdx.x * 256 + threadIdx.x;
  if (i < n) out[i] = (_Float16)in[i];
}

// Wq (H,D,DH) -> out[j*D + d] with j = h*DH+e  (n-major so B-frag k-pairs
// are contiguous dwords)
__global__ void cvt_w_qkv(const float* __restrict__ in,
                          _Float16* __restrict__ out) {
  int id = blockIdx.x * 256 + threadIdx.x;       // over D*D = 1M
  int j = id >> 10, d = id & 1023;
  out[(size_t)j * ATTN_D + d] =
      (_Float16)in[((size_t)(j >> 6) * ATTN_D + d) * ATTN_DH + (j & 63)];
}

// ---------------------------------------------------------------------------
// GEMM: C(8192 x 1024) = (A(8192 x 1024, f16 row-major) @ Bw + bias) * scale
// Bw is f16 stored [n*1024 + k] (n-major).
// MODE 0: store f16 at (B,H,T,DH)         (Q pre-scaled, K)
// MODE 1: store f16 at (B,H,DH,T)         (V transposed)
// MODE 2: store f32 row-major (B*T, D)    (final projection)
// block = 256 (8 waves); tile M=256 (32 rows/wave, 2 strips), N=64;
// K chunked by 32; B-tile staging software-pipelined through registers.
// ---------------------------------------------------------------------------
template <int MODE>
__global__ __launch_bounds__(256, 1)
void gemm_wmma(const _Float16* __restrict__ A,
               const _Float16* __restrict__ Bw,
               const float* __restrict__ bias, float scale,
               void* __restrict__ outp) {
  __shared__ _Float16 Bs[64 * 32];               // 4 KB, n-major [c][k]
  const int wave = threadIdx.x >> 5;
  const int lane = threadIdx.x & 31;
  const int lh = lane >> 4, l16 = lane & 15;
  const int mbase = blockIdx.x * 256 + wave * 32;
  const int nbase = blockIdx.y * 64;

  v8f acc[2][4];
#pragma unroll
  for (int s = 0; s < 2; ++s)
#pragma unroll
    for (int t = 0; t < 4; ++t)
      acc[s][t] = v8f{0.f, 0.f, 0.f, 0.f, 0.f, 0.f, 0.f, 0.f};

  const _Float16* Ar0 = A + (size_t)(mbase + l16) * ATTN_D;
  const _Float16* Ar1 = Ar0 + (size_t)16 * ATTN_D;

  // software-pipelined B staging: global -> reg (next) while LDS holds (cur)
  const int c = threadIdx.x >> 2;
  const int kk = (threadIdx.x & 3) * 8;
  const _Float16* Bsrc = Bw + (size_t)(nbase + c) * ATTN_D + kk;
  uint4 breg = *(const uint4*)Bsrc;

  for (int k0 = 0; k0 < ATTN_D; k0 += 32) {
    __syncthreads();                             // previous tile consumed
    *(uint4*)&Bs[c * 32 + kk] = breg;
    if (k0 + 32 < ATTN_D) breg = *(const uint4*)(Bsrc + k0 + 32);
    __syncthreads();

    v16h af[2];
    {
      v8u ua0, ua1;
#pragma unroll
      for (int j = 0; j < 8; ++j) {
        ua0[j] = *(const unsigned int*)(Ar0 + k0 + koffA(j, lh));
        ua1[j] = *(const unsigned int*)(Ar1 + k0 + koffA(j, lh));
      }
      af[0] = __builtin_bit_cast(v16h, ua0);
      af[1] = __builtin_bit_cast(v16h, ua1);
    }

#pragma unroll
    for (int t = 0; t < 4; ++t) {
      v8u ub;
#pragma unroll
      for (int j = 0; j < 8; ++j)
        ub[j] = *(const unsigned int*)&Bs[(t * 16 + l16) * 32 + koffB(j, lh)];
      v16h bf = __builtin_bit_cast(v16h, ub);
      acc[0][t] = __builtin_amdgcn_wmma_f32_16x16x32_f16(
          false, af[0], false, bf, (short)0, acc[0][t], false, false);
      acc[1][t] = __builtin_amdgcn_wmma_f32_16x16x32_f16(
          false, af[1], false, bf, (short)0, acc[1][t], false, false);
    }
  }

#pragma unroll
  for (int s = 0; s < 2; ++s) {
#pragma unroll
    for (int t = 0; t < 4; ++t) {
      int j = nbase + t * 16 + l16;
      float bj = bias[j];
#pragma unroll
      for (int i = 0; i < 8; ++i) {
        int r = mbase + s * 16 + i + 8 * lh;
        float val = (acc[s][t][i] + bj) * scale;
        if constexpr (MODE == 2) {
          ((float*)outp)[(size_t)r * ATTN_D + j] = val;
        } else {
          int b = r >> 11, tt = r & 2047, h = j >> 6, e = j & 63;
          size_t off = (MODE == 0)
              ? (((size_t)(b * ATTN_H + h) * ATTN_T + tt) * ATTN_DH + e)
              : (((size_t)(b * ATTN_H + h) * ATTN_DH + e) * ATTN_T + tt);
          ((_Float16*)outp)[off] = (_Float16)val;
        }
      }
    }
  }
}

// ---------------------------------------------------------------------------
// One 64-wide key block of flash attention for one wave (16 query rows).
// MASKED=false for interior blocks (kcol0+63 <= row0): no compares at all.
// ---------------------------------------------------------------------------
template <bool MASKED>
__device__ __forceinline__ void attn_block(
    int kcol0, int row0, int lh, int l16,
    const _Float16* __restrict__ Kb, const _Float16* __restrict__ Vb,
    _Float16* __restrict__ Pw,
    const v16h (&aq)[2], v8f (&co)[4], float (&rmax)[8], float (&rsum)[8]) {
  // ---- S = Q K^T (log2 units) -------------------------------------------
  v8f cs[4];
#pragma unroll
  for (int t = 0; t < 4; ++t) cs[t] = v8f{0.f, 0.f, 0.f, 0.f, 0.f, 0.f, 0.f, 0.f};
#pragma unroll
  for (int e0i = 0; e0i < 2; ++e0i) {
#pragma unroll
    for (int t = 0; t < 4; ++t) {
      v8u u;
#pragma unroll
      for (int j = 0; j < 8; ++j) {
        int tk = kcol0 + t * 16 + l16;           // key column for this lane
        u[j] = *(const unsigned int*)&Kb[(size_t)tk * ATTN_DH +
                                         e0i * 32 + koffB(j, lh)];
      }
      v16h bk = __builtin_bit_cast(v16h, u);
      cs[t] = __builtin_amdgcn_wmma_f32_16x16x32_f16(
          false, aq[e0i], false, bk, (short)0, cs[t], false, false);
    }
  }

  // ---- online softmax (log2 domain) ---------------------------------------
#pragma unroll
  for (int i = 0; i < 8; ++i) {
    float mi = NEG_BIG;
#pragma unroll
    for (int t = 0; t < 4; ++t) {
      float s = cs[t][i];
      if constexpr (MASKED) {
        const int row = row0 + i + 8 * lh;
        int col = kcol0 + t * 16 + l16;
        if (col > row) s = NEG_BIG;              // causal mask (tail block)
        cs[t][i] = s;
      }
      mi = fmaxf(mi, s);
    }
#pragma unroll
    for (int off = 1; off < 16; off <<= 1)       // row max across 16-lane half
      mi = fmaxf(mi, __shfl_xor(mi, off, 32));
    float mnew = fmaxf(rmax[i], mi);
    float alpha = exp2f(rmax[i] - mnew);
    float ps = 0.f;
#pragma unroll
    for (int t = 0; t < 4; ++t) {
      float p = exp2f(cs[t][i] - mnew);
      cs[t][i] = p;
      ps += p;
    }
#pragma unroll
    for (int off = 1; off < 16; off <<= 1)
      ps += __shfl_xor(ps, off, 32);
    rsum[i] = rsum[i] * alpha + ps;
    rmax[i] = mnew;
#pragma unroll
    for (int t = 0; t < 4; ++t) co[t][i] *= alpha;
  }

  // ---- P: C-layout -> A-layout via wave-private LDS tile ------------------
#pragma unroll
  for (int t = 0; t < 4; ++t)
#pragma unroll
    for (int i = 0; i < 8; ++i)
      Pw[(i + 8 * lh) * 64 + t * 16 + l16] = (_Float16)cs[t][i];
  // same-wave DS ops are in order; region is wave-private -> no barrier

  // ---- O += P @ V ---------------------------------------------------------
#pragma unroll
  for (int k0i = 0; k0i < 2; ++k0i) {
    v8u up;
#pragma unroll
    for (int j = 0; j < 8; ++j)
      up[j] = *(const unsigned int*)&Pw[l16 * 64 + k0i * 32 + koffA(j, lh)];
    v16h ap = __builtin_bit_cast(v16h, up);
#pragma unroll
    for (int t = 0; t < 4; ++t) {
      v8u uv;
#pragma unroll
      for (int j = 0; j < 8; ++j) {
        int e = t * 16 + l16;                    // head-dim column
        uv[j] = *(const unsigned int*)&Vb[(size_t)e * ATTN_T + kcol0 +
                                          k0i * 32 + koffB(j, lh)];
      }
      v16h bv = __builtin_bit_cast(v16h, uv);
      co[t] = __builtin_amdgcn_wmma_f32_16x16x32_f16(
          false, ap, false, bv, (short)0, co[t], false, false);
    }
  }
}

// ---------------------------------------------------------------------------
// Flash attention, causal.  grid = (B*H, T/128), block = 256 (8 waves).
// Q pre-scaled by (1/sqrt(DH))*log2(e); only the final key block is masked.
// Q,K: (B,H,T,DH) f16.  Vt: (B,H,DH,T) f16.  O: (B*T, D) f16 (concat layout).
// ---------------------------------------------------------------------------
__global__ __launch_bounds__(256, 1)
void flash_attn(const _Float16* __restrict__ Q,
                const _Float16* __restrict__ K,
                const _Float16* __restrict__ Vt,
                _Float16* __restrict__ O) {
  __shared__ _Float16 Pl[8 * 16 * 64];           // 16 KB, wave-private tiles
  const int bh = blockIdx.x;
  const int qb = blockIdx.y;
  const int wave = threadIdx.x >> 5;
  const int lane = threadIdx.x & 31;
  const int lh = lane >> 4, l16 = lane & 15;
  _Float16* Pw = &Pl[wave * 16 * 64];

  const _Float16* Qb = Q + (size_t)bh * ATTN_T * ATTN_DH;
  const _Float16* Kb = K + (size_t)bh * ATTN_T * ATTN_DH;
  const _Float16* Vb = Vt + (size_t)bh * ATTN_DH * ATTN_T;

  const int row0 = qb * 128 + wave * 16;         // first query row of wave

  // Q A-fragments over the DH (=64) contraction dim: 2 chunks of 32
  v16h aq[2];
#pragma unroll
  for (int e0i = 0; e0i < 2; ++e0i) {
    v8u u;
#pragma unroll
    for (int j = 0; j < 8; ++j)
      u[j] = *(const unsigned int*)&Qb[(size_t)(row0 + l16) * ATTN_DH +
                                       e0i * 32 + koffA(j, lh)];
    aq[e0i] = __builtin_bit_cast(v16h, u);
  }

  float rmax[8], rsum[8];
  v8f co[4];
#pragma unroll
  for (int i = 0; i < 8; ++i) { rmax[i] = NEG_BIG; rsum[i] = 0.f; }
#pragma unroll
  for (int t = 0; t < 4; ++t) co[t] = v8f{0.f, 0.f, 0.f, 0.f, 0.f, 0.f, 0.f, 0.f};

  const int nkb = (row0 >> 6) + 1;               // key blocks (64 wide)

  // interior blocks: fully below the diagonal -> no masking
  for (int kb = 0; kb < nkb - 1; ++kb) {
    const int kcol0 = kb * 64;
    {   // prefetch next key block's K rows and V columns (128 B per row)
      const _Float16* kpf = &Kb[(size_t)(kcol0 + 64 + lane) * ATTN_DH];
      __builtin_prefetch(kpf, 0, 3);
      __builtin_prefetch(kpf + (size_t)32 * ATTN_DH, 0, 3);
      const _Float16* vpf = &Vb[(size_t)lane * ATTN_T + kcol0 + 64];
      __builtin_prefetch(vpf, 0, 3);
      __builtin_prefetch(vpf + (size_t)32 * ATTN_T, 0, 3);
    }
    attn_block<false>(kcol0, row0, lh, l16, Kb, Vb, Pw, aq, co, rmax, rsum);
  }
  // final (diagonal) block: causal-masked
  attn_block<true>((nkb - 1) * 64, row0, lh, l16, Kb, Vb, Pw, aq, co, rmax, rsum);

  // ---- epilogue: normalize, store concat layout (B*T, D) ------------------
  const int b = bh >> 4, h = bh & 15;
#pragma unroll
  for (int t = 0; t < 4; ++t) {
    int e = t * 16 + l16;
#pragma unroll
    for (int i = 0; i < 8; ++i) {
      int row = row0 + i + 8 * lh;
      float val = co[t][i] / rsum[i];
      O[(size_t)(b * ATTN_T + row) * ATTN_D + h * ATTN_DH + e] = (_Float16)val;
    }
  }
}

// ---------------------------------------------------------------------------
// launch
// ---------------------------------------------------------------------------
extern "C" void kernel_launch(void* const* d_in, const int* in_sizes, int n_in,
                              void* d_out, int out_size, void* d_ws, size_t ws_size,
                              hipStream_t stream) {
  const float* x  = (const float*)d_in[0];
  // d_in[1] = mask (causal tril) -- computed analytically, unused
  const float* Wq = (const float*)d_in[2];
  const float* bq = (const float*)d_in[3];
  const float* Wk = (const float*)d_in[4];
  const float* bk = (const float*)d_in[5];
  const float* Wv = (const float*)d_in[6];
  const float* bv = (const float*)d_in[7];
  const float* Wo = (const float*)d_in[8];
  const float* bo = (const float*)d_in[9];

  char* ws = (char*)d_ws;
  const size_t MB = 1u << 20;
  _Float16* xh  = (_Float16*)(ws);              // 16 MB  (8192x1024)
  _Float16* wqT = (_Float16*)(ws + 16 * MB);    //  2 MB
  _Float16* wkT = (_Float16*)(ws + 18 * MB);    //  2 MB
  _Float16* wvT = (_Float16*)(ws + 20 * MB);    //  2 MB
  _Float16* woH = (_Float16*)(ws + 22 * MB);    //  2 MB
  _Float16* Qh  = (_Float16*)(ws + 24 * MB);    // 16 MB  (B,H,T,DH)
  _Float16* Kh  = (_Float16*)(ws + 40 * MB);    // 16 MB  (B,H,T,DH)
  _Float16* Vth = (_Float16*)(ws + 56 * MB);    // 16 MB  (B,H,DH,T)
  _Float16* Oh  = (_Float16*)(ws + 72 * MB);    // 16 MB  (B*T, D)

  const int nX = ATTN_BT * ATTN_D;              // 8388608
  cvt_f32_to_f16<<<(nX + 255) / 256, 256, 0, stream>>>(x, xh, nX);
  cvt_w_qkv<<<(ATTN_D * ATTN_D) / 256, 256, 0, stream>>>(Wq, wqT);
  cvt_w_qkv<<<(ATTN_D * ATTN_D) / 256, 256, 0, stream>>>(Wk, wkT);
  cvt_w_qkv<<<(ATTN_D * ATTN_D) / 256, 256, 0, stream>>>(Wv, wvT);
  cvt_f32_to_f16<<<(ATTN_D * ATTN_D) / 256, 256, 0, stream>>>(Wo, woH,
                                                              ATTN_D * ATTN_D);

  const float qscale = 0.125f * LOG2E;          // 1/sqrt(DH) * log2(e)
  dim3 ggrid(ATTN_BT / 256, ATTN_D / 64);       // (32, 16)
  gemm_wmma<0><<<ggrid, 256, 0, stream>>>(xh, wqT, bq, qscale, (void*)Qh);
  gemm_wmma<0><<<ggrid, 256, 0, stream>>>(xh, wkT, bk, 1.0f, (void*)Kh);
  gemm_wmma<1><<<ggrid, 256, 0, stream>>>(xh, wvT, bv, 1.0f, (void*)Vth);

  dim3 agrid(ATTN_B * ATTN_H, ATTN_T / 128);    // (64, 16)
  flash_attn<<<agrid, 256, 0, stream>>>(Qh, Kh, Vth, Oh);

  gemm_wmma<2><<<ggrid, 256, 0, stream>>>(Oh, woH, bo, 1.0f, d_out);
}
